// VectorQuantizer_77309411657
// MI455X (gfx1250) — matrix operand order repaired and checked by reading, not compiled, not run on hardware
//
#include <hip/hip_runtime.h>
#include <hip/hip_bf16.h>

// ---------------------------------------------------------------------------
// VQ: z (32,256,64,64) fp32, codebook (1024,256) fp32.
// N = 131072 rows, E_DIM = 256, N_E = 1024.
// Outputs (flat, fp32): z_q (33554432) | codebook_loss (1) | cls_loss (1) |
//                       indices (131072)
// ---------------------------------------------------------------------------

typedef __attribute__((ext_vector_type(16))) __bf16 v16bf;
typedef __attribute__((ext_vector_type(8)))  float  v8f;

#define NROWS   131072          // B*H*W
#define NE      1024
#define ED      256
#define ZQ_ELEMS 33554432ull    // B*C*H*W
#define LOSS_OFF 33554432ull
#define CLS_OFF  33554433ull
#define IDX_OFF  33554434ull

static __device__ __forceinline__ __bf16 f2bf(float f) {
    // round-to-nearest-even fp32 -> bf16
    unsigned u = __builtin_bit_cast(unsigned, f);
    u += 0x7FFFu + ((u >> 16) & 1u);
    unsigned short h = (unsigned short)(u >> 16);
    return __builtin_bit_cast(__bf16, h);
}

// ---- Kernel 1: codebook -> bf16 copy + fp32 row norms ---------------------
__global__ void vq_prep(const float* __restrict__ cb,
                        __bf16* __restrict__ cbb,
                        float* __restrict__ cbn) {
    int j = blockIdx.x;          // codebook row
    int t = threadIdx.x;         // 0..255 (channel)
    float v = cb[(size_t)j * ED + t];
    cbb[(size_t)j * ED + t] = f2bf(v);
    float s = v * v;
    #pragma unroll
    for (int off = 16; off > 0; off >>= 1)
        s += __shfl_xor(s, off, 32);
    __shared__ float red[8];
    if ((t & 31) == 0) red[t >> 5] = s;
    __syncthreads();
    if (t == 0) {
        float tot = 0.f;
        #pragma unroll
        for (int i = 0; i < 8; ++i) tot += red[i];
        cbn[j] = tot;
    }
}

// ---- Kernel 2: zero the loss accumulators ---------------------------------
__global__ void vq_zero(float* __restrict__ out) {
    out[LOSS_OFF] = 0.0f;
    out[CLS_OFF]  = 0.0f;
}

// ---- Kernel 3: WMMA argmin over codebook ----------------------------------
// One wave handles 16 consecutive z rows (positions along W, coalesced).
// score(m, j) = ||e_j||^2 - 2 * <z_m, e_j>   (||z||^2 is row-constant).
__global__ void __launch_bounds__(128)
vq_argmin(const float* __restrict__ z,
          const __bf16* __restrict__ cbb,
          const float* __restrict__ cbn,
          int* __restrict__ idxout) {
    const int lane    = threadIdx.x & 31;
    const int wave    = threadIdx.x >> 5;
    const int rowBase = (blockIdx.x * 4 + wave) * 16;

    // --- preload A panel: 16 rows x 256 K, bf16, lanes 0-15 K=0..15, lanes 16-31 K=16..31
    const int m   = lane & 15;
    const int row = rowBase + m;              // flat (b*H + h)*W + w
    const int b   = row >> 12;
    const int hw  = row & 4095;
    const float* zrow = z + ((size_t)b << 20) + (size_t)hw;  // + K*4096
    const int kbaseA = (lane & 16) ? 16 : 0;

    v16bf A[8];
    #pragma unroll
    for (int kt = 0; kt < 8; ++kt) {
        #pragma unroll
        for (int e = 0; e < 16; ++e) {
            int K = kt * 32 + kbaseA + e;
            A[kt][e] = f2bf(zrow[(size_t)K << 12]);
        }
    }

    // --- B layout: lane holds column n = lane&15; lanes 0-15 K=0..15,
    //     lanes 16-31 K=16..31 within each 32-deep chunk.
    const int nn      = lane & 15;
    const int kHalfB  = (lane & 16) ? 16 : 0;

    float minv[8];
    int   mini[8];
    #pragma unroll
    for (int r = 0; r < 8; ++r) { minv[r] = 3.4e38f; mini[r] = 0; }

    for (int jt = 0; jt < 64; ++jt) {
        const int j = jt * 16 + nn;
        const __bf16* brow = cbb + (size_t)j * ED + kHalfB;

        v8f acc = {0.f, 0.f, 0.f, 0.f, 0.f, 0.f, 0.f, 0.f};
        #pragma unroll
        for (int kt = 0; kt < 8; ++kt) {
            v16bf Bv = *(const v16bf*)(brow + kt * 32);
            acc = __builtin_amdgcn_wmma_f32_16x16x32_bf16(
                /*neg_a=*/false, A[kt], /*neg_b=*/false, Bv,
                /*c_mod=*/(short)0, acc, /*reuse_a=*/false, /*reuse_b=*/false);
        }

        const float cn = cbn[j];
        #pragma unroll
        for (int r = 0; r < 8; ++r) {
            float s = __builtin_fmaf(-2.0f, acc[r], cn);
            if (s < minv[r]) { minv[r] = s; mini[r] = j; }
        }
    }

    // --- cross-lane (16-lane halves) min+argmin reduction, ties -> low index
    #pragma unroll
    for (int r = 0; r < 8; ++r) {
        float v = minv[r];
        int   i = mini[r];
        #pragma unroll
        for (int off = 1; off < 16; off <<= 1) {
            float ov = __shfl_xor(v, off, 32);
            int   oi = __shfl_xor(i, off, 32);
            if (ov < v || (ov == v && oi < i)) { v = ov; i = oi; }
        }
        minv[r] = v; mini[r] = i;
    }

    // lanes 0-15 carry rows 0-7 (VGPR r -> M=r); lanes 16-31 rows 8-15
    if (lane == 0) {
        #pragma unroll
        for (int r = 0; r < 8; ++r) idxout[rowBase + r] = mini[r];
    } else if (lane == 16) {
        #pragma unroll
        for (int r = 0; r < 8; ++r) idxout[rowBase + 8 + r] = mini[r];
    }
}

// ---- Kernel 4: gather z_q (BCHW), MSE loss, float indices -----------------
__global__ void vq_gather(const float* __restrict__ z,
                          const float* __restrict__ cb,
                          const int* __restrict__ idx,
                          float* __restrict__ out) {
    const int n0 = blockIdx.x * 64;       // 64 positions per block (one W row)
    const int t  = threadIdx.x;           // 256 threads
    const int p  = t & 63;
    const int c0 = t >> 6;                // 0..3
    const int n  = n0 + p;
    const int id = idx[n];
    const int b  = n >> 12;
    const int hw = n & 4095;
    const float* zp  = z   + ((size_t)b << 20) + (size_t)hw;
    float*       op  = out + ((size_t)b << 20) + (size_t)hw;
    const float* cbp = cb  + (size_t)id * ED;

    float lsum = 0.f;
    #pragma unroll 4
    for (int c = c0; c < ED; c += 4) {
        float v  = cbp[c];
        float zv = zp[(size_t)c << 12];
        float d  = v - zv;
        lsum += d * d;
        op[(size_t)c << 12] = v;
    }

    if (t < 64) out[IDX_OFF + (size_t)(n0 + t)] = (float)idx[n0 + t];

    #pragma unroll
    for (int off = 16; off > 0; off >>= 1)
        lsum += __shfl_xor(lsum, off, 32);
    __shared__ float red[8];
    if ((t & 31) == 0) red[t >> 5] = lsum;
    __syncthreads();
    if (t == 0) {
        float tot = 0.f;
        #pragma unroll
        for (int i = 0; i < 8; ++i) tot += red[i];
        atomicAdd(out + LOSS_OFF, tot * (1.0f / 33554432.0f));
    }
}

// ---------------------------------------------------------------------------
extern "C" void kernel_launch(void* const* d_in, const int* in_sizes, int n_in,
                              void* d_out, int out_size, void* d_ws, size_t ws_size,
                              hipStream_t stream) {
    const float* z  = (const float*)d_in[0];   // (32,256,64,64)
    const float* cb = (const float*)d_in[1];   // (1024,256)
    float* out = (float*)d_out;

    // workspace layout
    __bf16* cbb = (__bf16*)d_ws;                                   // 512 KB
    float*  cbn = (float*)((char*)d_ws + 524288);                  //   4 KB
    int*    idx = (int*)((char*)d_ws + 524288 + 4096);             // 512 KB

    vq_prep<<<NE, ED, 0, stream>>>(cb, cbb, cbn);
    vq_zero<<<1, 1, 0, stream>>>(out);
    vq_argmin<<<NROWS / 64, 128, 0, stream>>>(z, cbb, cbn, idx);   // 4 waves/block
    vq_gather<<<NROWS / 64, 256, 0, stream>>>(z, cb, idx, out);
}
